// MultiHeadAttention_73151882985937
// MI455X (gfx1250) — compile-verified, matrix operand order checked by reading, not run on hardware
//
#include <hip/hip_runtime.h>
#include <hip/hip_bf16.h>
#include <math.h>

// Problem constants (fixed by the reference)
#define B_   2
#define S_   4096
#define D_   512
#define H_   8
#define HD_  64
#define SCALE_ 0.125f   // 1/sqrt(64)
#define M_   (B_ * S_)  // 8192 rows for all projections

typedef __attribute__((ext_vector_type(16))) __bf16 v16bf;
typedef __attribute__((ext_vector_type(8)))  float  v8f;

#ifndef __has_builtin
#define __has_builtin(x) 0
#endif
#if __has_builtin(__builtin_amdgcn_tensor_load_to_lds) && \
    __has_builtin(__builtin_amdgcn_s_wait_tensorcnt)
#define HAVE_TDM 1
#else
#define HAVE_TDM 0
#endif

struct __attribute__((aligned(16))) U4 { unsigned int x, y, z, w; };
union FragU { v16bf v; U4 q[2]; };

__device__ __forceinline__ __bf16 to_bf16(float f) {
  union { float f; unsigned u; } a; a.f = f;
  unsigned r = a.u + 0x7FFFu + ((a.u >> 16) & 1u);   // round-to-nearest-even
  union { unsigned short s; __bf16 b; } o; o.s = (unsigned short)(r >> 16);
  return o.b;
}

// WMMA bf16 fragment load. CDNA5 16-bit A 16x32 layout (CDNA5 ISA):
// lane l -> row (l&15); elements 0..7 = K = (l>>4)*8 .. +7 ; elements 8..15 = K+16..
// All B operands are pre-transposed so their fragments are the same pattern:
// two contiguous 16-byte chunks per lane (global_load_b128 / ds_load_b128).
__device__ __forceinline__ v16bf load_frag(const __bf16* base,
                                           int ld, int row0, int k0) {
  const int lane = threadIdx.x & 31;
  const __bf16* p = base + (size_t)(row0 + (lane & 15)) * ld
                         + (k0 + ((lane >> 4) << 3));
  FragU f;
  f.q[0] = *(const U4*)(p);
  f.q[1] = *(const U4*)(p + 16);
  return f.v;
}

__device__ __forceinline__ v8f wmma_bf16(v16bf a, v16bf b, v8f c) {
  // (neg_a, A, neg_b, B, c_mod, C, reuse_a, reuse_b)
  return __builtin_amdgcn_wmma_f32_16x16x32_bf16(false, a, false, b,
                                                 (short)0, c, false, false);
}

#if HAVE_TDM
typedef __attribute__((ext_vector_type(4))) unsigned int u32x4;
typedef __attribute__((ext_vector_type(4))) int          i32x4;
typedef __attribute__((ext_vector_type(8))) int          i32x8;

// Low 32 bits of a generic pointer to LDS are the LDS byte offset
// (ISA 10.2).
__device__ __forceinline__ unsigned lds_off_of(const void* p) {
  return (unsigned)(unsigned long long)p;
}

// Issue TENSOR_LOAD_TO_LDS for a 2D bf16 tile (tile_h rows x tile_w elems),
// source row stride = `stride` elements. D# packing per CDNA5 ISA:
//  g0: count=1 | lds_addr | global_addr[31:0] | global_addr[56:32] + type=2
//  g1: data_size=1(2B) @ [17:16]; tensor_dim0 @ [79:48]; tensor_dim1 @ [111:80];
//      tile_dim0 @ [127:112]; tile_dim1 @ [143:128]; dim0_stride @ [207:160]
__device__ __forceinline__ void tdm_load_2d(const void* gptr, unsigned lds_off,
                                            unsigned row_len, unsigned nrows,
                                            unsigned long long stride,
                                            unsigned tile_w, unsigned tile_h) {
  const unsigned long long ga = (unsigned long long)gptr;
  u32x4 g0;
  g0[0] = 1u;                                          // count=1, user mode
  g0[1] = lds_off;                                     // lds_addr
  g0[2] = (unsigned)(ga & 0xFFFFFFFFu);                // global_addr lo
  g0[3] = (unsigned)((ga >> 32) & 0x01FFFFFFu) | (2u << 30);  // hi + type=2
  i32x8 g1;
  g1[0] = (int)(1u << 16);                             // data_size = 2 bytes
  g1[1] = (int)((row_len & 0xFFFFu) << 16);            // tensor_dim0[15:0]
  g1[2] = (int)(((row_len >> 16) & 0xFFFFu) | ((nrows & 0xFFFFu) << 16));
  g1[3] = (int)(((nrows >> 16) & 0xFFFFu) | ((tile_w & 0xFFFFu) << 16));
  g1[4] = (int)(tile_h & 0xFFFFu);                     // tile_dim1 (tile_dim2=0)
  g1[5] = (int)(unsigned)(stride & 0xFFFFFFFFu);       // dim0_stride lo
  g1[6] = (int)(unsigned)((stride >> 32) & 0xFFFFu);   // dim0_stride hi
  g1[7] = 0;
  i32x4 z = {0, 0, 0, 0};
#if __clang_major__ >= 23
  i32x8 z8 = {};
  __builtin_amdgcn_tensor_load_to_lds(g0, g1, z, z, z8, 0);
#else
  __builtin_amdgcn_tensor_load_to_lds(g0, g1, z, z, 0);
#endif
}
#endif  // HAVE_TDM

// ---------------------------------------------------------------------------
// Elementwise fp32 -> bf16
__global__ __launch_bounds__(256)
void k_f32_to_bf16(const float* __restrict__ src, __bf16* __restrict__ dst, int n) {
  int i = blockIdx.x * 256 + threadIdx.x;
  if (i < n) dst[i] = to_bf16(src[i]);
}

// Transpose 512x512 weight [K][N] -> [N][K] with fp32->bf16 conversion
__global__ __launch_bounds__(256)
void k_transpose_w(const float* __restrict__ src, __bf16* __restrict__ dst) {
  int i = blockIdx.x * 256 + threadIdx.x;   // i over 512*512
  int k = i >> 9, n = i & 511;
  dst[(size_t)n * D_ + k] = to_bf16(src[i]);
}

// ---------------------------------------------------------------------------
// GEMM: C[M,N] = A[M,K] @ Bt[N,K]^T (+bias)*preScale, M=8192, N=K=512.
// 8 waves/block; all 8 waves share the same 32-row A strip, so the A K-chunk
// (32x32 bf16, 2KB) is staged into LDS by the TDM, double buffered.
// mode 1: bf16 out, head-split [B,H,S,Hd]   (Q, K projections)
// mode 2: bf16 out, head-split T [B,H,Hd,S] (V projection)
// mode 3: fp32 out, [M,N]                   (final projection)
__global__ __launch_bounds__(256)
void k_gemm_bf16(const __bf16* __restrict__ A, const __bf16* __restrict__ Bt,
                 const float* __restrict__ bias, void* __restrict__ dst,
                 float preScale, int mode) {
  const int lane = threadIdx.x & 31;
  const int wave = threadIdx.x >> 5;
  const int tile = blockIdx.x * 8 + wave;       // 4096 tiles total
  const int tilesN = D_ / 32;                   // 16
  const int m0 = (tile / tilesN) * 32;          // same for all waves in block
  const int n0 = (tile % tilesN) * 32;

#if HAVE_TDM
  __shared__ __align__(16) __bf16 As[2][32 * 32];
  if (wave == 0)
    tdm_load_2d(A + (size_t)m0 * D_, lds_off_of(As[0]), D_, M_, D_, 32, 32);
#endif

  v8f cc[2][2] = {};
  for (int k0 = 0; k0 < D_; k0 += 32) {
#if HAVE_TDM
    const int buf = (k0 >> 5) & 1;
    if (wave == 0) {
      if (k0 + 32 < D_) {
        tdm_load_2d(A + (size_t)m0 * D_ + (k0 + 32), lds_off_of(As[buf ^ 1]),
                    D_, M_, D_, 32, 32);
        __builtin_amdgcn_s_wait_tensorcnt(1);   // current chunk landed
      } else {
        __builtin_amdgcn_s_wait_tensorcnt(0);
      }
    }
    __syncthreads();
    v16bf a0 = load_frag(As[buf], 32, 0,  0);
    v16bf a1 = load_frag(As[buf], 32, 16, 0);
#else
    v16bf a0 = load_frag(A, D_, m0,      k0);
    v16bf a1 = load_frag(A, D_, m0 + 16, k0);
#endif
    v16bf b0 = load_frag(Bt, D_, n0,      k0);
    v16bf b1 = load_frag(Bt, D_, n0 + 16, k0);
    if (k0 + 32 < D_)   // next-chunk prefetch -> global_prefetch_b8
      __builtin_prefetch(Bt + (size_t)(n0 + (lane & 15)) * D_ + k0 + 32, 0, 3);
    cc[0][0] = wmma_bf16(a0, b0, cc[0][0]);
    cc[0][1] = wmma_bf16(a0, b1, cc[0][1]);
    cc[1][0] = wmma_bf16(a1, b0, cc[1][0]);
    cc[1][1] = wmma_bf16(a1, b1, cc[1][1]);
#if HAVE_TDM
    __syncthreads();    // all waves done with As[buf] before it is re-filled
#endif
  }

  const int nl = lane & 15, hi = lane >> 4;
  for (int j = 0; j < 2; ++j) {
    const int n = n0 + j * 16 + nl;
    const float bs = bias ? bias[n] : 0.0f;
    for (int i = 0; i < 2; ++i) {
      for (int r = 0; r < 8; ++r) {
        const int m = m0 + i * 16 + r + 8 * hi;          // C/D layout row
        const float val = (cc[i][j][r] + bs) * preScale;
        if (mode == 1) {
          const int b = m / S_, s = m % S_, h = n / HD_, d = n % HD_;
          ((__bf16*)dst)[(((size_t)b * H_ + h) * S_ + s) * HD_ + d] = to_bf16(val);
        } else if (mode == 2) {
          const int b = m / S_, s = m % S_, h = n / HD_, d = n % HD_;
          ((__bf16*)dst)[(((size_t)b * H_ + h) * HD_ + d) * S_ + s] = to_bf16(val);
        } else {
          ((float*)dst)[(size_t)m * D_ + n] = val;
        }
      }
    }
  }
}

// ---------------------------------------------------------------------------
// Flash attention. One wave owns a 16-query tile; the 4 waves of a block all
// process the SAME (b,h), so K/V chunks (32 keys) are staged once per block
// into LDS by the TDM (double buffered, 4KB+4KB per buffer), then consumed by
// all waves. Online softmax over key chunks of 32; scale pre-folded into Q;
// P bounced through LDS for C-layout -> A-fragment conversion.
__global__ __launch_bounds__(128)
void k_flash_attn(const __bf16* __restrict__ Q,  // [B,H,S,Hd] (pre-scaled)
                  const __bf16* __restrict__ K,  // [B,H,S,Hd]
                  const __bf16* __restrict__ Vt, // [B,H,Hd,S]
                  __bf16* __restrict__ Ab) {     // [B,S,D] head-interleaved
  const int wave = threadIdx.x >> 5;
  const int lane = threadIdx.x & 31;
  const int gw   = blockIdx.x * 4 + wave;        // 4096 waves total
  const int qt   = gw % (S_ / 16);
  const int bh   = gw / (S_ / 16);               // identical across the block
  const int h    = bh % H_, b = bh / H_;

  const __bf16* Qbh = Q  + (size_t)bh * S_ * HD_;
  const __bf16* Kbh = K  + (size_t)bh * S_ * HD_;
  const __bf16* Vbh = Vt + (size_t)bh * HD_ * S_;

  __shared__ __align__(16) __bf16 Pl[4][16 * 32];
#if HAVE_TDM
  __shared__ __align__(16) __bf16 Ks[2][32 * HD_];   // [keys][d]
  __shared__ __align__(16) __bf16 Vs[2][HD_ * 32];   // [d][keys]
  if (wave == 0) {
    tdm_load_2d(Kbh, lds_off_of(Ks[0]), HD_, S_, HD_, HD_, 32);
    tdm_load_2d(Vbh, lds_off_of(Vs[0]), S_, HD_, S_, 32, HD_);
  }
#endif
  __bf16* pl = Pl[wave];

  const v16bf qa0 = load_frag(Qbh, HD_, qt * 16, 0);
  const v16bf qa1 = load_frag(Qbh, HD_, qt * 16, 32);

  v8f o0 = {}, o1 = {}, o2 = {}, o3 = {};
  float mrun[8], lrun[8];
  for (int r = 0; r < 8; ++r) { mrun[r] = -3.0e38f; lrun[r] = 0.0f; }

  const int nl = lane & 15, hi = lane >> 4;

  for (int kc = 0; kc < S_; kc += 32) {
#if HAVE_TDM
    const int buf = (kc >> 5) & 1;
    if (wave == 0) {
      if (kc + 32 < S_) {
        tdm_load_2d(Kbh + (size_t)(kc + 32) * HD_, lds_off_of(Ks[buf ^ 1]),
                    HD_, S_, HD_, HD_, 32);
        tdm_load_2d(Vbh + (kc + 32), lds_off_of(Vs[buf ^ 1]),
                    S_, HD_, S_, 32, HD_);
        __builtin_amdgcn_s_wait_tensorcnt(2);   // current chunk's 2 ops landed
      } else {
        __builtin_amdgcn_s_wait_tensorcnt(0);
      }
    }
    __syncthreads();
    const __bf16* kp = Ks[buf];  const int kld = HD_;
    const __bf16* vp = Vs[buf];  const int vld = 32;
#else
    const __bf16* kp = Kbh + (size_t)kc * HD_;  const int kld = HD_;
    const __bf16* vp = Vbh + kc;                const int vld = S_;
#endif

    // scores: Q[16x64] x K^T[64x32] -> two 16x16 f32 tiles, K-dim chained 2x32
    v8f c0 = {}, c1 = {};
    c0 = wmma_bf16(qa0, load_frag(kp, kld, 0,  0),  c0);
    c0 = wmma_bf16(qa1, load_frag(kp, kld, 0,  32), c0);
    c1 = wmma_bf16(qa0, load_frag(kp, kld, 16, 0),  c1);
    c1 = wmma_bf16(qa1, load_frag(kp, kld, 16, 32), c1);

    // online softmax: rows live inside a 16-lane half -> shfl_xor width 16
    float fac[8];
    for (int r = 0; r < 8; ++r) {
      float mx = fmaxf(c0[r], c1[r]);
      for (int m = 1; m < 16; m <<= 1) mx = fmaxf(mx, __shfl_xor(mx, m, 16));
      const float nm = fmaxf(mrun[r], mx);
      const float p0 = __expf(c0[r] - nm);
      const float p1 = __expf(c1[r] - nm);
      c0[r] = p0; c1[r] = p1;
      float rs = p0 + p1;
      for (int m = 1; m < 16; m <<= 1) rs += __shfl_xor(rs, m, 16);
      fac[r]  = __expf(mrun[r] - nm);
      lrun[r] = lrun[r] * fac[r] + rs;
      mrun[r] = nm;
    }
    for (int r = 0; r < 8; ++r) {
      o0[r] *= fac[r]; o1[r] *= fac[r]; o2[r] *= fac[r]; o3[r] *= fac[r];
    }

    // P: C-layout -> LDS row-major [16 q][32 k] -> A-fragment
    for (int r = 0; r < 8; ++r) {
      pl[(r + 8 * hi) * 32 + nl]      = to_bf16(c0[r]);
      pl[(r + 8 * hi) * 32 + 16 + nl] = to_bf16(c1[r]);
    }
    __syncthreads();
    const v16bf pa = load_frag(pl, 32, 0, 0);
    o0 = wmma_bf16(pa, load_frag(vp, vld, 0,  0), o0);
    o1 = wmma_bf16(pa, load_frag(vp, vld, 16, 0), o1);
    o2 = wmma_bf16(pa, load_frag(vp, vld, 32, 0), o2);
    o3 = wmma_bf16(pa, load_frag(vp, vld, 48, 0), o3);
    __syncthreads();   // buffers free before wave0 refills them next iter
  }

  // epilogue: normalize and scatter into [B,S,D] (D = H*Hd interleaved)
  for (int r = 0; r < 8; ++r) {
    const float inv = 1.0f / lrun[r];
    const int q = qt * 16 + r + 8 * hi;
    const size_t rowoff = ((size_t)b * S_ + q) * D_ + h * HD_;
    Ab[rowoff +  0 + nl] = to_bf16(o0[r] * inv);
    Ab[rowoff + 16 + nl] = to_bf16(o1[r] * inv);
    Ab[rowoff + 32 + nl] = to_bf16(o2[r] * inv);
    Ab[rowoff + 48 + nl] = to_bf16(o3[r] * inv);
  }
}

// ---------------------------------------------------------------------------
extern "C" void kernel_launch(void* const* d_in, const int* in_sizes, int n_in,
                              void* d_out, int out_size, void* d_ws, size_t ws_size,
                              hipStream_t stream) {
  (void)in_sizes; (void)n_in; (void)out_size; (void)ws_size;
  const float* x  = (const float*)d_in[0];
  const float* wq = (const float*)d_in[1];
  const float* bq = (const float*)d_in[2];
  const float* wk = (const float*)d_in[3];
  const float* bk = (const float*)d_in[4];
  const float* wv = (const float*)d_in[5];
  const float* bv = (const float*)d_in[6];
  const float* wo = (const float*)d_in[7];
  const float* bo = (const float*)d_in[8];
  float* out = (float*)d_out;

  const size_t BSD = (size_t)B_ * S_ * D_;   // 4,194,304 elems
  const size_t WW  = (size_t)D_ * D_;        //   262,144 elems

  char* w = (char*)d_ws;
  __bf16* xb  = (__bf16*)w; w += BSD * 2;
  __bf16* wqT = (__bf16*)w; w += WW * 2;
  __bf16* wkT = (__bf16*)w; w += WW * 2;
  __bf16* wvT = (__bf16*)w; w += WW * 2;
  __bf16* woT = (__bf16*)w; w += WW * 2;
  __bf16* Qb  = (__bf16*)w; w += BSD * 2;
  __bf16* Kb  = (__bf16*)w; w += BSD * 2;
  __bf16* Vt  = (__bf16*)w; w += BSD * 2;
  __bf16* Ab  = (__bf16*)w; w += BSD * 2;   // ~44 MB total workspace

  // 1) precision conversion + weight transposes
  k_f32_to_bf16<<<(int)(BSD / 256), 256, 0, stream>>>(x, xb, (int)BSD);
  k_transpose_w<<<(int)(WW / 256), 256, 0, stream>>>(wq, wqT);
  k_transpose_w<<<(int)(WW / 256), 256, 0, stream>>>(wk, wkT);
  k_transpose_w<<<(int)(WW / 256), 256, 0, stream>>>(wv, wvT);
  k_transpose_w<<<(int)(WW / 256), 256, 0, stream>>>(wo, woT);

  // 2) Q/K/V projections (4096 wave-tiles -> 512 blocks of 8 waves)
  k_gemm_bf16<<<512, 256, 0, stream>>>(xb, wqT, bq, (void*)Qb, SCALE_, 1);
  k_gemm_bf16<<<512, 256, 0, stream>>>(xb, wkT, bk, (void*)Kb, 1.0f,   1);
  k_gemm_bf16<<<512, 256, 0, stream>>>(xb, wvT, bv, (void*)Vt, 1.0f,   2);

  // 3) flash attention: 4096 query-tile waves -> 1024 blocks of 4 waves
  k_flash_attn<<<1024, 128, 0, stream>>>(Qb, Kb, Vt, Ab);

  // 4) output projection, fp32 + bias
  k_gemm_bf16<<<512, 256, 0, stream>>>(Ab, woT, bo, (void*)out, 1.0f, 3);
}
